// STL_6459630813195
// MI455X (gfx1250) — matrix operand order repaired and checked by reading, not compiled, not run on hardware
//
#include <hip/hip_runtime.h>
#include <hip/hip_bf16.h>
#include <math.h>

// ---------------- CDNA5 WMMA plumbing ----------------
typedef __attribute__((ext_vector_type(16))) __bf16          v16bf;
typedef __attribute__((ext_vector_type(8)))  float           v8f;
typedef __attribute__((ext_vector_type(8)))  unsigned short  us8;   // 16 bytes

__device__ __forceinline__ unsigned short f2bfbits(float f) {
    union { float f; unsigned u; } a; a.f = f;
    unsigned u = a.u;
    unsigned r = u + 0x7FFFu + ((u >> 16) & 1u);   // round-to-nearest-even
    return (unsigned short)(r >> 16);
}

union FragU { us8 h[2]; v16bf v; };

// A-fragment: row-major 16-bit tile, K-range [kb, kb+32).  p = row + kb.
// lane half selects k in {mhalf*16..mhalf*16+15}, contiguous in k
__device__ __forceinline__ v16bf fragA(const unsigned short* p, int mhalf) {
    FragU u;
    u.h[0] = *(const us8*)(p + (mhalf << 4));
    u.h[1] = *(const us8*)(p + 8 + (mhalf << 4));
    return u.v;
}
// B-fragment: element e <-> k = kb + mhalf*16 + e, contiguous in k.
// p = row_of_n + kb + mhalf*16 (16B aligned)
__device__ __forceinline__ v16bf fragB(const unsigned short* p) {
    FragU u;
    u.h[0] = *(const us8*)(p);
    u.h[1] = *(const us8*)(p + 8);
    return u.v;
}

__device__ __forceinline__ v8f wmma_bf16(v16bf a, v16bf b, v8f c) {
    return __builtin_amdgcn_wmma_f32_16x16x32_bf16(false, a, false, b, (short)0, c, false, false);
}

__device__ __forceinline__ float wave_sum(float v) {
    for (int o = 16; o > 0; o >>= 1) v += __shfl_xor(v, o, 32);
    return v;
}
__device__ __forceinline__ float wave_max(float v) {
    for (int o = 16; o > 0; o >>= 1) v = fmaxf(v, __shfl_xor(v, o, 32));
    return v;
}

// pack 4 floats -> 4 bf16 -> one 8-byte store (p must be 8B aligned)
__device__ __forceinline__ void store_bf4(unsigned short* p, float a, float b, float c, float d) {
    unsigned u0 = (unsigned)f2bfbits(a) | ((unsigned)f2bfbits(b) << 16);
    unsigned u1 = (unsigned)f2bfbits(c) | ((unsigned)f2bfbits(d) << 16);
    *(uint2*)p = make_uint2(u0, u1);
}

// ---------------- problem constants ----------------
#define C_DIM 180
#define CPAD  192
#define NHEAD 6
#define HDIM  30
#define NTOK  64
#define SS_   4
#define NWIN  4096
#define QK_SCALE 0.18257418583505536f  // 30^-0.5

// =====================================================================
// Prep: zero-padded row-major bf16 weight copy  dst[Npad][Kpad]
// =====================================================================
__global__ __launch_bounds__(256) void k_padw(
    const float* __restrict__ src, unsigned short* __restrict__ dst,
    int N, int K, int Npad, int Kpad)
{
    int i = blockIdx.x * 256 + threadIdx.x;
    if (i >= Npad * Kpad) return;
    int n = i / Kpad, k = i - n * Kpad;
    float v = (n < N && k < K) ? src[n * K + k] : 0.f;
    dst[i] = f2bfbits(v);
}

// =====================================================================
// Kernel A: LN1 + roll(-4,-4) + partition + QKV GEMM (64x540, K=180->192)
// Q[wh][64][32] (scaled), K[wh][64][32], Vt[wh][32][64]  all bf16, pads zeroed
// =====================================================================
__global__ __launch_bounds__(256) void k_ln1_qkv(
    const float* __restrict__ x, const float* __restrict__ ln_g, const float* __restrict__ ln_b,
    const unsigned short* __restrict__ wq, const float* __restrict__ qkv_b,
    unsigned short* __restrict__ Qb, unsigned short* __restrict__ Kb,
    unsigned short* __restrict__ Vt)
{
    __shared__ __align__(16) unsigned short xs[NTOK * CPAD];
    const int w = blockIdx.x;
    const int b = w >> 8, wi = w & 255, wy = wi >> 4, wx = wi & 15;
    const int tid = threadIdx.x, lane = tid & 31, wave = tid >> 5;
    const int mhalf = lane >> 4;

    // zero hd=30,31 pads of Q/K cols and Vt rows for this window
    for (int i = tid; i < NHEAD * NTOK * 2; i += 256) {
        int h = i / (NTOK * 2), rem = i - h * (NTOK * 2), m = rem >> 1, p = 30 + (rem & 1);
        int wh = w * NHEAD + h;
        Qb[(wh * NTOK + m) * 32 + p] = 0;
        Kb[(wh * NTOK + m) * 32 + p] = 0;
        Vt[(wh * 32 + p) * NTOK + m] = 0;
    }

    // gather rolled window + LayerNorm in registers (wave -> 8 tokens)
    for (int tt = 0; tt < 8; ++tt) {
        int t = wave * 8 + tt;
        int ty = t >> 3, tx = t & 7;
        int y  = (wy * 8 + ty + SS_) & 127;
        int xx = (wx * 8 + tx + SS_) & 127;
        const float* row = x + (size_t)((b << 14) + (y << 7) + xx) * C_DIM;
        float4 v0 = ((const float4*)row)[lane];            // chunks 0..31
        float4 v1 = make_float4(0.f, 0.f, 0.f, 0.f);
        bool has2 = (lane < 13);                            // chunks 32..44
        if (has2) v1 = ((const float4*)row)[lane + 32];
        float s = v0.x + v0.y + v0.z + v0.w + v1.x + v1.y + v1.z + v1.w;
        float q = v0.x*v0.x + v0.y*v0.y + v0.z*v0.z + v0.w*v0.w
                + v1.x*v1.x + v1.y*v1.y + v1.z*v1.z + v1.w*v1.w;
        s = wave_sum(s); q = wave_sum(q);
        float mu = s * (1.f / C_DIM);
        float rs = rsqrtf(q * (1.f / C_DIM) - mu * mu + 1e-5f);
        float4 g0 = ((const float4*)ln_g)[lane];
        float4 b0 = ((const float4*)ln_b)[lane];
        store_bf4(xs + t * CPAD + lane * 4,
                  (v0.x - mu) * rs * g0.x + b0.x, (v0.y - mu) * rs * g0.y + b0.y,
                  (v0.z - mu) * rs * g0.z + b0.z, (v0.w - mu) * rs * g0.w + b0.w);
        if (has2) {
            float4 g1 = ((const float4*)ln_g)[lane + 32];
            float4 b1 = ((const float4*)ln_b)[lane + 32];
            store_bf4(xs + t * CPAD + (lane + 32) * 4,
                      (v1.x - mu) * rs * g1.x + b1.x, (v1.y - mu) * rs * g1.y + b1.y,
                      (v1.z - mu) * rs * g1.z + b1.z, (v1.w - mu) * rs * g1.w + b1.w);
        }
        if (lane < 12) xs[t * CPAD + C_DIM + lane] = 0;     // K pad cols
    }
    __syncthreads();

    // GEMM: 64 x 544 tiles of 16x16, K = 192 (6 steps), guard-free fragments
    for (int t = wave; t < 4 * 34; t += 8) {
        int mt = t / 34, nt = t - mt * 34;
        int n  = nt * 16 + (lane & 15);
        int am = mt * 16 + (lane & 15);
        const unsigned short* arow = xs + am * CPAD;
        const unsigned short* wrow = wq + n * CPAD;
        v8f acc = {0.f,0.f,0.f,0.f,0.f,0.f,0.f,0.f};
        #pragma unroll
        for (int ks = 0; ks < 6; ++ks) {
            int kb = ks * 32;
            v16bf a  = fragA(arow + kb, mhalf);
            v16bf bm = fragB(wrow + kb + (mhalf << 4));
            acc = wmma_bf16(a, bm, acc);
        }
        if (n < 3 * C_DIM) {
            float bias = qkv_b[n];
            int sec = n / C_DIM, cc = n - sec * C_DIM;
            int h = cc / HDIM, hd = cc - h * HDIM;
            int wh = w * NHEAD + h;
            #pragma unroll
            for (int r = 0; r < 8; ++r) {
                int m = mt * 16 + r + (mhalf << 3);
                float v = acc[r] + bias;
                if (sec == 0)      Qb[(wh * NTOK + m) * 32 + hd] = f2bfbits(v * QK_SCALE);
                else if (sec == 1) Kb[(wh * NTOK + m) * 32 + hd] = f2bfbits(v);
                else               Vt[(wh * 32 + hd) * NTOK + m] = f2bfbits(v);
            }
        }
    }
}

// =====================================================================
// Kernel B: per (window, head): Q@K^T + bias + mask, softmax, P@V
// attn output padded to stride 192 (pads zeroed by head==0 blocks)
// =====================================================================
__global__ __launch_bounds__(128) void k_attn(
    const unsigned short* __restrict__ Qb, const unsigned short* __restrict__ Kb,
    const unsigned short* __restrict__ Vt, const float* __restrict__ rpb,
    unsigned short* __restrict__ attn)
{
    __shared__ float sc[NTOK * NTOK];
    __shared__ __align__(16) unsigned short pb[NTOK * NTOK];
    const int wh = blockIdx.x;
    const int w = wh / NHEAD, head = wh - w * NHEAD;
    const int wi = w & 255, wy = wi >> 4, wx = wi & 15;
    const int tid = threadIdx.x, lane = tid & 31, mt = tid >> 5;
    const int mhalf = lane >> 4;

    if (head == 0) {   // zero channel pads 180..191 of this window's attn rows
        for (int i = tid; i < NTOK * (CPAD - C_DIM); i += 128)
            attn[(size_t)w * NTOK * CPAD + (i / 12) * CPAD + C_DIM + (i % 12)] = 0;
    }

    // Q fragment (K=32, single step)
    v16bf aQ = fragA(Qb + ((size_t)wh * NTOK + mt * 16 + (lane & 15)) * 32, mhalf);

    for (int nt = 0; nt < 4; ++nt) {
        int n = nt * 16 + (lane & 15);
        v16bf bK = fragB(Kb + ((size_t)wh * NTOK + n) * 32 + (mhalf << 4));
        v8f acc = {0.f,0.f,0.f,0.f,0.f,0.f,0.f,0.f};
        acc = wmma_bf16(aQ, bK, acc);
        int ty2 = n >> 3, tx2 = n & 7;
        int by = wy * 8 + ty2, bx = wx * 8 + tx2;
        int rk = ((by < 120) ? 0 : (by < 124 ? 1 : 2)) * 3 + ((bx < 120) ? 0 : (bx < 124 ? 1 : 2));
        #pragma unroll
        for (int r = 0; r < 8; ++r) {
            int m = mt * 16 + r + (mhalf << 3);
            int ty = m >> 3, tx = m & 7;
            int idx = (ty - ty2 + 7) * 15 + (tx - tx2 + 7);
            float v = acc[r] + rpb[idx * NHEAD + head];
            int ay = wy * 8 + ty, ax = wx * 8 + tx;
            int rq = ((ay < 120) ? 0 : (ay < 124 ? 1 : 2)) * 3 + ((ax < 120) ? 0 : (ax < 124 ? 1 : 2));
            if (rq != rk) v -= 100.f;
            sc[m * NTOK + n] = v;
        }
    }
    __syncthreads();
    for (int row = mt * 16; row < mt * 16 + 16; ++row) {
        float s0 = sc[row * 64 + lane], s1 = sc[row * 64 + 32 + lane];
        float mx = wave_max(fmaxf(s0, s1));
        float e0 = __expf(s0 - mx), e1 = __expf(s1 - mx);
        float inv = 1.f / wave_sum(e0 + e1);
        pb[row * 64 + lane]      = f2bfbits(e0 * inv);
        pb[row * 64 + 32 + lane] = f2bfbits(e1 * inv);
    }
    __syncthreads();
    // P @ V (K = 64 -> 2 steps)
    for (int nt = 0; nt < 2; ++nt) {
        int n = nt * 16 + (lane & 15);
        v8f acc = {0.f,0.f,0.f,0.f,0.f,0.f,0.f,0.f};
        #pragma unroll
        for (int s = 0; s < 2; ++s) {
            v16bf aP = fragA(pb + (mt * 16 + (lane & 15)) * 64 + s * 32, mhalf);
            v16bf bV = fragB(Vt + ((size_t)wh * 32 + n) * 64 + s * 32 + (mhalf << 4));
            acc = wmma_bf16(aP, bV, acc);
        }
        if (n < HDIM) {
            #pragma unroll
            for (int r = 0; r < 8; ++r) {
                int m = mt * 16 + r + (mhalf << 3);
                attn[((size_t)w * NTOK + m) * CPAD + head * HDIM + n] = f2bfbits(acc[r]);
            }
        }
    }
}

// =====================================================================
// Kernel C: proj GEMM (64x180, K=192) + combine + roll(+4,+4) + residual
// =====================================================================
__global__ __launch_bounds__(256) void k_proj(
    const unsigned short* __restrict__ attn, const unsigned short* __restrict__ wp,
    const float* __restrict__ proj_b, const float* __restrict__ x,
    float* __restrict__ hout)
{
    __shared__ __align__(16) unsigned short as_[NTOK * CPAD];
    const int w = blockIdx.x;
    const int b = w >> 8, wi = w & 255, wy = wi >> 4, wx = wi & 15;
    const int tid = threadIdx.x, lane = tid & 31, wave = tid >> 5;
    const int mhalf = lane >> 4;

    const us8* src8 = (const us8*)(attn + (size_t)w * NTOK * CPAD);
    for (int i = tid; i < NTOK * CPAD / 8; i += 256) ((us8*)as_)[i] = src8[i];
    __syncthreads();

    for (int t = wave; t < 4 * 12; t += 8) {
        int mt = t / 12, nt = t - mt * 12;
        int n  = nt * 16 + (lane & 15);
        int am = mt * 16 + (lane & 15);
        const unsigned short* arow = as_ + am * CPAD;
        const unsigned short* wrow = wp + n * CPAD;
        v8f acc = {0.f,0.f,0.f,0.f,0.f,0.f,0.f,0.f};
        #pragma unroll
        for (int ks = 0; ks < 6; ++ks) {
            int kb = ks * 32;
            acc = wmma_bf16(fragA(arow + kb, mhalf), fragB(wrow + kb + (mhalf << 4)), acc);
        }
        if (n < C_DIM) {
            float bias = proj_b[n];
            #pragma unroll
            for (int r = 0; r < 8; ++r) {
                int m = mt * 16 + r + (mhalf << 3);
                int ty = m >> 3, tx = m & 7;
                int y  = (wy * 8 + ty + SS_) & 127;
                int xx = (wx * 8 + tx + SS_) & 127;
                int gi = ((b << 14) + (y << 7) + xx) * C_DIM + n;
                hout[gi] = x[gi] + acc[r] + bias;
            }
        }
    }
}

// =====================================================================
// Kernel D: LN2 + fc1 GEMM (64x360, K=192) + exact GELU -> m1 (stride 384)
// =====================================================================
__global__ __launch_bounds__(256) void k_mlp1(
    const float* __restrict__ h, const float* __restrict__ ln_g, const float* __restrict__ ln_b,
    const unsigned short* __restrict__ w1, const float* __restrict__ fc1_b,
    unsigned short* __restrict__ m1)
{
    __shared__ __align__(16) unsigned short xs[NTOK * CPAD];
    const int t0 = blockIdx.x * NTOK;
    const int tid = threadIdx.x, lane = tid & 31, wave = tid >> 5;
    const int mhalf = lane >> 4;

    // zero m1 channel pads 360..383
    for (int i = tid; i < NTOK * 24; i += 256)
        m1[(size_t)(t0 + i / 24) * 384 + 360 + (i % 24)] = 0;

    for (int tt = 0; tt < 8; ++tt) {
        int t = wave * 8 + tt;
        const float* row = h + (size_t)(t0 + t) * C_DIM;
        float4 v0 = ((const float4*)row)[lane];
        float4 v1 = make_float4(0.f, 0.f, 0.f, 0.f);
        bool has2 = (lane < 13);
        if (has2) v1 = ((const float4*)row)[lane + 32];
        float s = v0.x + v0.y + v0.z + v0.w + v1.x + v1.y + v1.z + v1.w;
        float q = v0.x*v0.x + v0.y*v0.y + v0.z*v0.z + v0.w*v0.w
                + v1.x*v1.x + v1.y*v1.y + v1.z*v1.z + v1.w*v1.w;
        s = wave_sum(s); q = wave_sum(q);
        float mu = s * (1.f / C_DIM);
        float rs = rsqrtf(q * (1.f / C_DIM) - mu * mu + 1e-5f);
        float4 g0 = ((const float4*)ln_g)[lane];
        float4 b0 = ((const float4*)ln_b)[lane];
        store_bf4(xs + t * CPAD + lane * 4,
                  (v0.x - mu) * rs * g0.x + b0.x, (v0.y - mu) * rs * g0.y + b0.y,
                  (v0.z - mu) * rs * g0.z + b0.z, (v0.w - mu) * rs * g0.w + b0.w);
        if (has2) {
            float4 g1 = ((const float4*)ln_g)[lane + 32];
            float4 b1 = ((const float4*)ln_b)[lane + 32];
            store_bf4(xs + t * CPAD + (lane + 32) * 4,
                      (v1.x - mu) * rs * g1.x + b1.x, (v1.y - mu) * rs * g1.y + b1.y,
                      (v1.z - mu) * rs * g1.z + b1.z, (v1.w - mu) * rs * g1.w + b1.w);
        }
        if (lane < 12) xs[t * CPAD + C_DIM + lane] = 0;
    }
    __syncthreads();

    for (int t = wave; t < 4 * 23; t += 8) {
        int mt = t / 23, nt = t - mt * 23;
        int n  = nt * 16 + (lane & 15);
        int am = mt * 16 + (lane & 15);
        const unsigned short* arow = xs + am * CPAD;
        const unsigned short* wrow = w1 + n * CPAD;
        v8f acc = {0.f,0.f,0.f,0.f,0.f,0.f,0.f,0.f};
        #pragma unroll
        for (int ks = 0; ks < 6; ++ks) {
            int kb = ks * 32;
            acc = wmma_bf16(fragA(arow + kb, mhalf), fragB(wrow + kb + (mhalf << 4)), acc);
        }
        if (n < 2 * C_DIM) {
            float bias = fc1_b[n];
            #pragma unroll
            for (int r = 0; r < 8; ++r) {
                int m = mt * 16 + r + (mhalf << 3);
                float v = acc[r] + bias;
                v = 0.5f * v * (1.f + erff(v * 0.70710678118654752f));
                m1[(size_t)(t0 + m) * 384 + n] = f2bfbits(v);
            }
        }
    }
}

// =====================================================================
// Kernel E: fc2 GEMM (64x180, K=384) + residual, in place on d_out
// =====================================================================
__global__ __launch_bounds__(256) void k_mlp2(
    const unsigned short* __restrict__ m1, const unsigned short* __restrict__ w2,
    const float* __restrict__ fc2_b, float* out)
{
    __shared__ __align__(16) unsigned short as_[NTOK * 384];
    const int t0 = blockIdx.x * NTOK;
    const int tid = threadIdx.x, lane = tid & 31, wave = tid >> 5;
    const int mhalf = lane >> 4;

    const us8* src8 = (const us8*)(m1 + (size_t)t0 * 384);
    for (int i = tid; i < NTOK * 384 / 8; i += 256) ((us8*)as_)[i] = src8[i];
    __syncthreads();

    for (int t = wave; t < 4 * 12; t += 8) {
        int mt = t / 12, nt = t - mt * 12;
        int n  = nt * 16 + (lane & 15);
        int am = mt * 16 + (lane & 15);
        const unsigned short* arow = as_ + am * 384;
        const unsigned short* wrow = w2 + n * 384;
        v8f acc = {0.f,0.f,0.f,0.f,0.f,0.f,0.f,0.f};
        #pragma unroll
        for (int ks = 0; ks < 12; ++ks) {
            int kb = ks * 32;
            acc = wmma_bf16(fragA(arow + kb, mhalf), fragB(wrow + kb + (mhalf << 4)), acc);
        }
        if (n < C_DIM) {
            float bias = fc2_b[n];
            #pragma unroll
            for (int r = 0; r < 8; ++r) {
                int m = mt * 16 + r + (mhalf << 3);
                int gi = (t0 + m) * C_DIM + n;
                out[gi] = out[gi] + acc[r] + bias;
            }
        }
    }
}

// =====================================================================
extern "C" void kernel_launch(void* const* d_in, const int* in_sizes, int n_in,
                              void* d_out, int out_size, void* d_ws, size_t ws_size,
                              hipStream_t stream) {
    const float* x      = (const float*)d_in[0];
    const float* ln1_g  = (const float*)d_in[1];
    const float* ln1_b  = (const float*)d_in[2];
    const float* qkv_w  = (const float*)d_in[3];
    const float* qkv_b  = (const float*)d_in[4];
    const float* proj_w = (const float*)d_in[5];
    const float* proj_b = (const float*)d_in[6];
    const float* rpb    = (const float*)d_in[7];
    const float* ln2_g  = (const float*)d_in[8];
    const float* ln2_b  = (const float*)d_in[9];
    const float* fc1_w  = (const float*)d_in[10];
    const float* fc1_b  = (const float*)d_in[11];
    const float* fc2_w  = (const float*)d_in[12];
    const float* fc2_b  = (const float*)d_in[13];

    unsigned short* ws = (unsigned short*)d_ws;
    unsigned short* wq = ws;                   // 544*192
    unsigned short* wp = wq + 544 * 192;       // 192*192
    unsigned short* w1 = wp + 192 * 192;       // 368*192
    unsigned short* w2 = w1 + 368 * 192;       // 192*384
    unsigned short* dyn = w2 + 192 * 384;      // == ws + 285,696 (16B-aligned)

    const size_t QKV_ELEMS = (size_t)NWIN * NHEAD * NTOK * 32;   // 50,331,648
    unsigned short* Qb   = dyn;
    unsigned short* Kb   = Qb + QKV_ELEMS;
    unsigned short* Vt   = Kb + QKV_ELEMS;
    unsigned short* attn = Vt + QKV_ELEMS;                       // 4096*64*192
    unsigned short* m1   = Qb;                                   // reuse dead Q+K region
    float* hbuf = (float*)d_out;

    k_padw<<<(544 * 192 + 255) / 256, 256, 0, stream>>>(qkv_w,  wq, 540, 180, 544, 192);
    k_padw<<<(192 * 192 + 255) / 256, 256, 0, stream>>>(proj_w, wp, 180, 180, 192, 192);
    k_padw<<<(368 * 192 + 255) / 256, 256, 0, stream>>>(fc1_w,  w1, 360, 180, 368, 192);
    k_padw<<<(192 * 384 + 255) / 256, 256, 0, stream>>>(fc2_w,  w2, 180, 360, 192, 384);

    k_ln1_qkv<<<NWIN, 256, 0, stream>>>(x, ln1_g, ln1_b, wq, qkv_b, Qb, Kb, Vt);
    k_attn   <<<NWIN * NHEAD, 128, 0, stream>>>(Qb, Kb, Vt, rpb, attn);
    k_proj   <<<NWIN, 256, 0, stream>>>(attn, wp, proj_b, x, hbuf);
    k_mlp1   <<<NWIN, 256, 0, stream>>>(hbuf, ln2_g, ln2_b, w1, fc1_b, m1);
    k_mlp2   <<<NWIN, 256, 0, stream>>>(m1, w2, fc2_b, hbuf);
}
